// MaskedXAttn_3186865734404
// MI455X (gfx1250) — matrix operand order repaired and checked by reading, not compiled, not run on hardware
//
#include <hip/hip_runtime.h>
#include <hip/hip_bf16.h>

// ---- MI455X / gfx1250 masked cross-attention, WMMA f16 (f32 accum) ----
// B=4, NQ=100, HW=4096, D=1024, H=16, HD=64

typedef _Float16 h16;
typedef __attribute__((ext_vector_type(16))) _Float16 v16h;
typedef __attribute__((ext_vector_type(8)))  _Float16 v8h;
typedef __attribute__((ext_vector_type(8)))  float    v8f;

#define NB   4
#define NQ   100
#define HWK  4096
#define DIM  1024
#define NH   16
#define HD   64
#define NEGV (-1e9f)

// ---------------------------------------------------------------- convert
__global__ __launch_bounds__(256)
void cvt_f32_f16(const float* __restrict__ x, h16* __restrict__ y, int n) {
    int i = blockIdx.x * 256 + threadIdx.x;
    if (i < n) y[i] = (h16)x[i];
}

// ---------------------------------------------------------------- layernorm
// one 256-thread block per row of 1024; f32 in -> f16 out
__global__ __launch_bounds__(256)
void ln_kernel(const float* __restrict__ x, const float* __restrict__ g,
               const float* __restrict__ bt, h16* __restrict__ y) {
    int row = blockIdx.x;
    int tid = threadIdx.x;
    const float* xr = x + (size_t)row * DIM;
    float v[4]; float s = 0.f, s2 = 0.f;
#pragma unroll
    for (int i = 0; i < 4; i++) { float t = xr[tid * 4 + i]; v[i] = t; s += t; s2 += t * t; }
#pragma unroll
    for (int d = 1; d < 32; d <<= 1) { s += __shfl_xor(s, d, 32); s2 += __shfl_xor(s2, d, 32); }
    __shared__ float sm[8], sm2[8];
    if ((tid & 31) == 0) { sm[tid >> 5] = s; sm2[tid >> 5] = s2; }
    __syncthreads();
    float ts = 0.f, ts2 = 0.f;
#pragma unroll
    for (int w = 0; w < 8; w++) { ts += sm[w]; ts2 += sm2[w]; }
    float mean = ts * (1.f / DIM);
    float var  = ts2 * (1.f / DIM) - mean * mean;
    float rstd = rsqrtf(var + 1e-5f);
#pragma unroll
    for (int i = 0; i < 4; i++) {
        int idx = tid * 4 + i;
        y[(size_t)row * DIM + idx] = (h16)((v[i] - mean) * rstd * g[idx] + bt[idx]);
    }
}

// ---------------------------------------------------------------- mask prep
// one 128-thread block per (b,q) row. Packs 4096 int32 -> 128 u32 bit-words,
// folding in the "no allowed key -> attend everywhere" rule (all bits set).
__global__ __launch_bounds__(128)
void maskprep_kernel(const int* __restrict__ mask, unsigned* __restrict__ effbits) {
    int row = blockIdx.x;                   // b*NQ + q
    int t = threadIdx.x;                    // 0..127, one u32 word each
    const int* mr = mask + (size_t)row * HWK + t * 32;
    unsigned w = 0;
#pragma unroll
    for (int i = 0; i < 32; i++) if (mr[i]) w |= (1u << i);
    int any = (w != 0);
#pragma unroll
    for (int d = 1; d < 32; d <<= 1) any |= __shfl_xor(any, d, 32);
    __shared__ int sm[4];
    if ((t & 31) == 0) sm[t >> 5] = any;
    __syncthreads();
    int tot = sm[0] | sm[1] | sm[2] | sm[3];
    effbits[(size_t)row * 128 + t] = tot ? w : 0xFFFFFFFFu;
}

// ---------------------------------------------------------------- WMMA GEMM
// C[m,n] = sum_k X[m,k] * W[n,k] + bias[n]    (K = 1024 fixed)
// one wave computes a 32x64 tile (2 row-halves share every B load):
//   grid.x = ceil(M/32), grid.y = 16 (N groups of 64)
// mode 0: f16 out[m*1024+n]   mode 1: f32 out[m*1024+n]
// mode 2: f16 V-transposed out[((b*16+h)*64+hd)*4096 + key], m=(b,key), n=(h,hd)
__global__ __launch_bounds__(32)
void gemm32(const h16* __restrict__ X, const h16* __restrict__ W,
            const float* __restrict__ bias, int M,
            h16* __restrict__ outH, float* __restrict__ outF, int mode) {
    const int mt = blockIdx.x, ng = blockIdx.y;
    const int lane = threadIdx.x, hid = lane >> 4, l16 = lane & 15;

    int m0 = mt * 32 + l16;      if (m0 >= M) m0 = M - 1;
    int m1 = mt * 32 + 16 + l16; if (m1 >= M) m1 = M - 1;
    const h16* xr0 = X + (size_t)m0 * DIM;
    const h16* xr1 = X + (size_t)m1 * DIM;

    v8f acc0[4] = {}, acc1[4] = {};
    for (int kb = 0; kb < DIM; kb += 32) {
        // A operands: lane l16 = row, elements 0..15 -> k = kb+16*hid+e
        v16h a0, a1;
        {
            v8h lo = *(const v8h*)(xr0 + kb + 16 * hid);
            v8h hi = *(const v8h*)(xr0 + kb + 16 * hid + 8);
#pragma unroll
            for (int i = 0; i < 8; i++) { a0[i] = lo[i]; a0[i + 8] = hi[i]; }
        }
        {
            v8h lo = *(const v8h*)(xr1 + kb + 16 * hid);
            v8h hi = *(const v8h*)(xr1 + kb + 16 * hid + 8);
#pragma unroll
            for (int i = 0; i < 8; i++) { a1[i] = lo[i]; a1[i + 8] = hi[i]; }
        }
#pragma unroll
        for (int s = 0; s < 4; s++) {
            int n = (ng * 4 + s) * 16 + l16;
            v16h bm = *(const v16h*)(W + (size_t)n * DIM + kb + 16 * hid);
            acc0[s] = __builtin_amdgcn_wmma_f32_16x16x32_f16(
                false, a0, false, bm, (short)0, acc0[s], false, false);
            acc1[s] = __builtin_amdgcn_wmma_f32_16x16x32_f16(
                false, a1, false, bm, (short)0, acc1[s], false, false);
        }
    }
#pragma unroll
    for (int s = 0; s < 4; s++) {
        int n = (ng * 4 + s) * 16 + l16;
        float bv = bias ? bias[n] : 0.f;
#pragma unroll
        for (int half = 0; half < 2; half++) {
#pragma unroll
            for (int r = 0; r < 8; r++) {
                int m = mt * 32 + half * 16 + r + 8 * hid;
                if (m < M) {
                    float v = (half ? acc1[s][r] : acc0[s][r]) + bv;
                    if (mode == 0)      outH[(size_t)m * DIM + n] = (h16)v;
                    else if (mode == 1) outF[(size_t)m * DIM + n] = v;
                    else {
                        int b = m >> 12, key = m & 4095, hh = n >> 6, hd = n & 63;
                        outH[(((size_t)(b * NH + hh) * HD + hd) * HWK) + key] = (h16)v;
                    }
                }
            }
        }
    }
}

// ---------------------------------------------------------------- fused attention
// one wave per (qtile, h, b); flash-style over HW in steps of 32 keys.
// qp: [B][NQ][D] f16, kp: [B][HW][D] f16, vt: [B][H][HD][HW] f16
// effbits: [B][NQ][128] u32 (bit=1 -> key allowed), ctx: [B][NQ][D] f16
__global__ __launch_bounds__(32)
void attn_kernel(const h16* __restrict__ qp, const h16* __restrict__ kp,
                 const h16* __restrict__ vt, const unsigned* __restrict__ effbits,
                 h16* __restrict__ ctx) {
    const int qt = blockIdx.x, h = blockIdx.y, b = blockIdx.z;
    const int lane = threadIdx.x, hid = lane >> 4, l16 = lane & 15;

    __shared__ h16 sP[16][32];

    // ---- Q A-operands (hd chunks 0..31, 32..63), row = qt*16 + l16 (clamped)
    int qrowA = qt * 16 + l16; if (qrowA >= NQ) qrowA = NQ - 1;
    const h16* qbase = qp + ((size_t)(b * NQ + qrowA)) * DIM + h * HD;
    v16h qa[2];
#pragma unroll
    for (int c = 0; c < 2; c++) {
        v8h lo = *(const v8h*)(qbase + 32 * c + 16 * hid);
        v8h hi = *(const v8h*)(qbase + 32 * c + 16 * hid + 8);
#pragma unroll
        for (int i = 0; i < 8; i++) { qa[c][i] = lo[i]; qa[c][i + 8] = hi[i]; }
    }

    // per-lane rows: float index r <-> query row qt*16 + r + 8*hid
    float mrun[8], lrun[8];
    const unsigned* ebq[8];
#pragma unroll
    for (int r = 0; r < 8; r++) {
        mrun[r] = -1e30f; lrun[r] = 0.f;
        int q = qt * 16 + r + 8 * hid; if (q >= NQ) q = NQ - 1;
        ebq[r] = effbits + ((size_t)(b * NQ + q)) * 128;
    }

    v8f acc[4] = {};

    for (int kb = 0; kb < HWK; kb += 32) {
        // ---- scores: two 16-key subtiles, K=64 chained as 2x 16x16x32
        v8f s[2];
#pragma unroll
        for (int j = 0; j < 2; j++) {
            int key = kb + j * 16 + l16;
            const h16* kpp = kp + ((size_t)(b * HWK + key)) * DIM + h * HD + 16 * hid;
            v16h bk0 = *(const v16h*)(kpp);        // hd 0..31 chunk
            v16h bk1 = *(const v16h*)(kpp + 32);   // hd 32..63 chunk
            v8f sc = {};
            sc = __builtin_amdgcn_wmma_f32_16x16x32_f16(false, qa[0], false, bk0,
                                                        (short)0, sc, false, false);
            sc = __builtin_amdgcn_wmma_f32_16x16x32_f16(false, qa[1], false, bk1,
                                                        (short)0, sc, false, false);
            s[j] = sc;
        }
        const int wi = kb >> 5;
        // ---- mask + scale + online softmax (rows live on 16-lane halves)
#pragma unroll
        for (int r = 0; r < 8; r++) {
            unsigned w = ebq[r][wi];
            float s0 = ((w >> l16) & 1u)        ? s[0][r] * 0.125f : NEGV;
            float s1 = ((w >> (16 + l16)) & 1u) ? s[1][r] * 0.125f : NEGV;
            float mx = fmaxf(s0, s1);
#pragma unroll
            for (int d = 1; d < 16; d <<= 1) mx = fmaxf(mx, __shfl_xor(mx, d, 16));
            float mnew = fmaxf(mrun[r], mx);
            float corr = __expf(mrun[r] - mnew);
            float p0 = __expf(s0 - mnew), p1 = __expf(s1 - mnew);
            float ps = p0 + p1;
#pragma unroll
            for (int d = 1; d < 16; d <<= 1) ps += __shfl_xor(ps, d, 16);
            lrun[r] = lrun[r] * corr + ps;
            mrun[r] = mnew;
            acc[0][r] *= corr; acc[1][r] *= corr; acc[2][r] *= corr; acc[3][r] *= corr;
            int rr = r + 8 * hid;
            sP[rr][l16]      = (h16)p0;
            sP[rr][16 + l16] = (h16)p1;
        }
        __syncthreads();
        // ---- P: C-layout -> A-layout via LDS
        v16h pa;
#pragma unroll
        for (int e = 0; e < 16; e++) {
            int kk = e + 16 * hid;
            pa[e] = sP[l16][kk];
        }
        __syncthreads();
        // ---- ctx += P @ V  (4 hd chunks, V^T layout gives contiguous B loads)
#pragma unroll
        for (int c = 0; c < 4; c++) {
            const h16* vpp = vt + (((size_t)(b * NH + h) * HD + 16 * c + l16)) * HWK
                             + kb + 16 * hid;
            v16h vb = *(const v16h*)vpp;
            acc[c] = __builtin_amdgcn_wmma_f32_16x16x32_f16(
                false, pa, false, vb, (short)0, acc[c], false, false);
        }
    }
    // ---- normalize + store ctx (f16)
#pragma unroll
    for (int r = 0; r < 8; r++) {
        int q = qt * 16 + r + 8 * hid;
        if (q < NQ) {
            float inv = 1.f / lrun[r];
#pragma unroll
            for (int c = 0; c < 4; c++)
                ctx[((size_t)(b * NQ + q)) * DIM + h * HD + 16 * c + l16]
                    = (h16)(acc[c][r] * inv);
        }
    }
}

// ---------------------------------------------------------------- launch
extern "C" void kernel_launch(void* const* d_in, const int* in_sizes, int n_in,
                              void* d_out, int out_size, void* d_ws, size_t ws_size,
                              hipStream_t stream) {
    const float* q    = (const float*)d_in[0];
    const float* kv   = (const float*)d_in[1];
    const int*   mask = (const int*)d_in[2];
    const float* ipw  = (const float*)d_in[3];
    const float* ipb  = (const float*)d_in[4];
    const float* ow   = (const float*)d_in[5];
    const float* ob   = (const float*)d_in[6];
    const float* gq   = (const float*)d_in[7];
    const float* bq   = (const float*)d_in[8];
    const float* gkv  = (const float*)d_in[9];
    const float* bkv  = (const float*)d_in[10];
    float* out = (float*)d_out;

    char* ws = (char*)d_ws;
    size_t off = 0;
    auto take = [&](size_t bytes) -> char* {
        char* p = ws + off;
        off += (bytes + 255) & ~(size_t)255;
        return p;
    };
    const size_t QROWS = (size_t)NB * NQ;        // 400
    const size_t KROWS = (size_t)NB * HWK;       // 16384
    h16* qn   = (h16*)take(QROWS * DIM * 2);
    h16* kvn  = (h16*)take(KROWS * DIM * 2);
    h16* qp   = (h16*)take(QROWS * DIM * 2);
    h16* kp   = (h16*)take(KROWS * DIM * 2);
    h16* vt   = (h16*)take(KROWS * DIM * 2);     // [B][H][HD][HW]
    h16* cx   = (h16*)take(QROWS * DIM * 2);
    h16* w3   = (h16*)take((size_t)3 * DIM * DIM * 2);
    h16* ow16 = (h16*)take((size_t)DIM * DIM * 2);
    unsigned* ebits = (unsigned*)take(QROWS * 128 * 4);

    // weight conversion
    {
        int n = 3 * DIM * DIM;
        cvt_f32_f16<<<(n + 255) / 256, 256, 0, stream>>>(ipw, w3, n);
        n = DIM * DIM;
        cvt_f32_f16<<<(n + 255) / 256, 256, 0, stream>>>(ow, ow16, n);
    }
    // layernorms
    ln_kernel<<<(int)QROWS, 256, 0, stream>>>(q, gq, bq, qn);
    ln_kernel<<<(int)KROWS, 256, 0, stream>>>(kv, gkv, bkv, kvn);
    // mask -> effective bitmask (has_any rule folded in)
    maskprep_kernel<<<(int)QROWS, 128, 0, stream>>>(mask, ebits);

    // projections: qp = qn @ Wq^T + bq ; kp = kvn @ Wk^T + bk ; vt = (kvn @ Wv^T + bv)^T
    {
        dim3 gq_(13, 16);     // ceil(400/32), 1024/64
        gemm32<<<gq_, 32, 0, stream>>>(qn, w3, ipb, (int)QROWS, qp, nullptr, 0);
        dim3 gk_(512, 16);    // 16384/32
        gemm32<<<gk_, 32, 0, stream>>>(kvn, w3 + (size_t)DIM * DIM, ipb + DIM,
                                       (int)KROWS, kp, nullptr, 0);
        gemm32<<<gk_, 32, 0, stream>>>(kvn, w3 + (size_t)2 * DIM * DIM, ipb + 2 * DIM,
                                       (int)KROWS, vt, nullptr, 2);
    }
    // fused attention
    {
        dim3 ga(7, NH, NB);   // ceil(100/16)=7 q-tiles
        attn_kernel<<<ga, 32, 0, stream>>>(qp, kp, vt, ebits, cx);
    }
    // output projection: out = ctx @ out_w^T + out_b (f32)
    {
        dim3 go(13, 16);
        gemm32<<<go, 32, 0, stream>>>(cx, ow16, ob, (int)QROWS, nullptr, out, 1);
    }
    (void)in_sizes; (void)n_in; (void)out_size; (void)ws_size;
}
